// LlamaAttention_59725815218336
// MI455X (gfx1250) — matrix operand order repaired and checked by reading, not compiled, not run on hardware
//
#include <hip/hip_runtime.h>
#include <hip/hip_bf16.h>

#define B_    2
#define S_    2048
#define HID_  2048
#define NH_   16
#define NKV_  4
#define D_    128
#define NREP_ 4
#define NB_   2048
#define SCALING_ 0.08838834764831843f
#define NEG_INF_ (-1.0e9f)

typedef __attribute__((ext_vector_type(16))) _Float16 v16h;
typedef __attribute__((ext_vector_type(8)))  _Float16 v8h;
typedef __attribute__((ext_vector_type(8)))  float    v8f;
typedef __attribute__((ext_vector_type(4)))  unsigned u32x4;
typedef __attribute__((ext_vector_type(8)))  unsigned u32x8;

#define WMMA_F32F16(a, b, c) \
  __builtin_amdgcn_wmma_f32_16x16x32_f16(false, (a), false, (b), (short)0, (c), false, false)

static __device__ inline v16h make16(v8h a, v8h b) {
  v16h r;
#pragma unroll
  for (int i = 0; i < 8; ++i) { r[i] = a[i]; r[i + 8] = b[i]; }
  return r;
}

// A-fragment (16xK tile, row-major, per ISA 7.12.2): lanes 0-15 -> K 0..7 & 16..23,
// lanes 16-31 -> K 8..15 & 24..31, row = lane&15.
static __device__ inline v16h frag_a(const _Float16* base, int stride, int kc, int lane) {
  int row  = lane & 15;
  int koff = (lane < 16) ? 0 : 8;
  v8h lo = *(const v8h*)(base + row * stride + kc + koff);
  v8h hi = *(const v8h*)(base + row * stride + kc + 16 + koff);
  return make16(lo, hi);
}

// B-fragment (NxK strip stored row-major by N): lanes 0-15 -> K 0..7 & 16..23,
// lanes 16-31 -> K 8..15 & 24..31, col = lane&15.
static __device__ inline v16h frag_b(const _Float16* base, int stride, int kc, int lane) {
  int col  = lane & 15;
  int koff = (lane < 16) ? 0 : 8;
  v8h lo = *(const v8h*)(base + col * stride + kc + koff);
  v8h hi = *(const v8h*)(base + col * stride + kc + 16 + koff);
  return make16(lo, hi);
}

// Async global -> LDS (ASYNCcnt path, per-lane addresses).
static __device__ inline void async_b128_to_lds(const void* gp, unsigned lds) {
  asm volatile("global_load_async_to_lds_b128 %0, %1, off"
               :: "v"(lds), "v"(gp) : "memory");
}
static __device__ inline void wait_asynccnt0() {
  asm volatile("s_wait_asynccnt 0x0" ::: "memory");
}

// Tensor Data Mover: one-shot 2D tile load (TENSORcnt path).
// Elements are dwords (data_size=4B). td0/td1 = tensor dims, tile0/tile1 = tile dims,
// stride0 = tensor dim0 stride (dwords). D# packing per ISA 08_async_tensor 8.3/8.4.
static __device__ inline void tdm_load_2d(const void* gp, unsigned lds,
                                          unsigned td0, unsigned td1,
                                          unsigned tile0, unsigned tile1,
                                          unsigned stride0) {
  unsigned long long ga = (unsigned long long)(size_t)gp;
  u32x4 g0;
  g0[0] = 1u;                                              // count=1, user mode
  g0[1] = lds;                                             // lds_addr (bytes)
  g0[2] = (unsigned)ga;                                    // global_addr[31:0]
  g0[3] = (unsigned)((ga >> 32) & 0x01FFFFFFu) | 0x80000000u; // addr[56:32] | type=2
  u32x8 g1;
  g1[0] = (2u << 16);                                      // wg_mask=0, data_size=4B
  g1[1] = (td0 & 0xFFFFu) << 16;                           // tensor_dim0 lo16
  g1[2] = (td0 >> 16) | ((td1 & 0xFFFFu) << 16);           // td0 hi16 | td1 lo16
  g1[3] = (td1 >> 16) | (tile0 << 16);                     // td1 hi16 | tile_dim0
  g1[4] = tile1;                                           // tile_dim1 (tile_dim2=0)
  g1[5] = stride0;                                         // dim0 stride lo32
  g1[6] = 0u;
  g1[7] = 0u;
  asm volatile("tensor_load_to_lds %0, %1" :: "s"(g0), "s"(g1) : "memory");
}

// ---------------- conversion kernels ----------------
__global__ void k_cvt16(const float* __restrict__ in, _Float16* __restrict__ out, int n) {
  int i = blockIdx.x * 256 + threadIdx.x;
  if (i < n) out[i] = (_Float16)in[i];
}

// out[n*K + k] = (f16) in[k*N + n]   (weight transpose to (N,K) row-major)
__global__ void k_cvtT(const float* __restrict__ in, _Float16* __restrict__ out, int K, int N) {
  int i = blockIdx.x * 256 + threadIdx.x;
  if (i < N * K) {
    int n = i / K, k = i - n * K;
    out[i] = (_Float16)in[(size_t)k * N + n];
  }
}

__global__ void k_zero(unsigned* __restrict__ p, int n) {
  int i = blockIdx.x * 256 + threadIdx.x;
  if (i < n) p[i] = 0u;
}

// ---------------- WMMA GEMM: C(MxN,f32) = A(MxK,f16 rowmajor) * Bt(NxK,f16 rowmajor)^T ----
__global__ void __launch_bounds__(128)
k_gemm(const _Float16* __restrict__ A, const _Float16* __restrict__ Bt,
       float* __restrict__ C, int M, int N, int K) {
  __shared__ _Float16 As[64 * 32];
  __shared__ _Float16 Bs[64 * 32];
  const int tid  = threadIdx.x;
  const int lane = tid & 31, wv = tid >> 5;
  const int n0 = blockIdx.x * 64, m0 = blockIdx.y * 64;
  (void)M;
  v8f zero = {};
  v8f acc[4] = {zero, zero, zero, zero};

  for (int c = 0; c < K; c += 32) {
    __syncthreads();
    for (int i = tid; i < 256; i += 128) {
      int row = i >> 2, ch = (i & 3) * 8;
      async_b128_to_lds(A  + (size_t)(m0 + row) * K + c + ch,
                        (unsigned)(size_t)(As + row * 32 + ch));
      async_b128_to_lds(Bt + (size_t)(n0 + row) * K + c + ch,
                        (unsigned)(size_t)(Bs + row * 32 + ch));
      if (c + 32 < K) {
        __builtin_prefetch(A  + (size_t)(m0 + row) * K + c + 32 + ch, 0, 1);
        __builtin_prefetch(Bt + (size_t)(n0 + row) * K + c + 32 + ch, 0, 1);
      }
    }
    wait_asynccnt0();
    __syncthreads();
    v16h a = frag_a(As + wv * 16 * 32, 32, 0, lane);
#pragma unroll
    for (int j = 0; j < 4; ++j) {
      v16h b = frag_b(Bs + j * 16 * 32, 32, 0, lane);
      acc[j] = WMMA_F32F16(a, b, acc[j]);
    }
  }
  const int lh = lane >> 4, cg = lane & 15;
#pragma unroll
  for (int j = 0; j < 4; ++j)
#pragma unroll
    for (int r = 0; r < 8; ++r)
      C[(size_t)(m0 + wv * 16 + r + 8 * lh) * N + n0 + j * 16 + cg] = acc[j][r];
}

// ---------------- RoPE (+absmax) ----------------
__global__ void __launch_bounds__(128)
k_rope(const float* __restrict__ proj, const float* __restrict__ cosb,
       const float* __restrict__ sinb, float* __restrict__ o32,
       _Float16* __restrict__ o16, unsigned* __restrict__ absmax, int nheads) {
  int d = threadIdx.x;
  int s = blockIdx.x % S_;
  int h = (blockIdx.x / S_) % nheads;
  int b = blockIdx.x / (S_ * nheads);
  size_t prow = ((size_t)(b * S_ + s)) * (nheads * D_) + h * D_;
  float v    = proj[prow + d];
  int   dd   = (d < 64) ? d + 64 : d - 64;
  float part = proj[prow + dd];
  float sign = (d < 64) ? -1.f : 1.f;
  size_t crow = ((size_t)(b * S_ + s)) * D_;
  float r = v * cosb[crow + d] + sign * part * sinb[crow + d];
  size_t oi = ((size_t)((b * nheads + h) * S_ + s)) * D_ + d;
  o32[oi] = r;
  o16[oi] = (_Float16)r;
  __shared__ float red[128];
  red[threadIdx.x] = fabsf(r);
  __syncthreads();
  for (int st = 64; st > 0; st >>= 1) {
    if (threadIdx.x < st) red[threadIdx.x] = fmaxf(red[threadIdx.x], red[threadIdx.x + st]);
    __syncthreads();
  }
  if (threadIdx.x == 0) atomicMax(&absmax[h], __float_as_uint(red[0]));
}

// v: (b,s,kh,d) f32 -> transposed f16 (b,kh,d,s)
__global__ void k_vprep(const float* __restrict__ vproj, _Float16* __restrict__ vT) {
  int i = blockIdx.x * 256 + threadIdx.x;
  if (i >= B_ * NKV_ * D_ * S_) return;
  int s  = i % S_;
  int d  = (i / S_) % D_;
  int kh = (i / (S_ * D_)) % NKV_;
  int b  = i / (S_ * D_ * NKV_);
  vT[i] = (_Float16)vproj[((size_t)(b * S_ + s)) * (NKV_ * D_) + kh * D_ + d];
}

// ---------------- quantile machinery ----------------
__global__ void k_hist(const float* __restrict__ x, const unsigned* __restrict__ absmax,
                       unsigned* __restrict__ hist, int nheads) {
  int i = blockIdx.x * 256 + threadIdx.x;
  if (i >= B_ * nheads * S_ * D_) return;
  int h = (i / (S_ * D_)) % nheads;
  float am = __uint_as_float(absmax[h]);
  float a  = fabsf(x[i]);
  int bin = 0;
  if (am > 0.f) {
    bin = (int)(a / am * (float)NB_);
    if (bin > NB_ - 1) bin = NB_ - 1;
  }
  atomicAdd(&hist[h * NB_ + bin], 1u);
}

__global__ void k_scales(const unsigned* __restrict__ hq, const unsigned* __restrict__ aq,
                         const unsigned* __restrict__ hk, const unsigned* __restrict__ ak,
                         float* __restrict__ qs, float* __restrict__ ks,
                         float* __restrict__ qs_out, float* __restrict__ ks_out) {
  __shared__ float ksh[NKV_];
  int t = threadIdx.x;
  const unsigned total  = (unsigned)(B_ * S_ * D_);
  const unsigned target = (unsigned)(0.995f * (float)total + 0.5f);
  if (t < NH_) {
    float am = __uint_as_float(aq[t]);
    unsigned cum = 0; int bin = NB_ - 1;
    for (int i = 0; i < NB_; ++i) { cum += hq[t * NB_ + i]; if (cum >= target) { bin = i; break; } }
    float sc = fmaxf(am * (float)(bin + 1) / (float)NB_, 1e-6f) / 7.0f;
    qs[t] = sc; qs_out[t] = sc;
  } else if (t < NH_ + NKV_) {
    int kh = t - NH_;
    float am = __uint_as_float(ak[kh]);
    unsigned cum = 0; int bin = NB_ - 1;
    for (int i = 0; i < NB_; ++i) { cum += hk[kh * NB_ + i]; if (cum >= target) { bin = i; break; } }
    float sc = fmaxf(am * (float)(bin + 1) / (float)NB_, 1e-6f) / 7.0f;
    ks[kh] = sc; ksh[kh] = sc;
  }
  __syncthreads();
  if (t < NH_) ks_out[t] = ksh[t / NREP_];
}

__global__ void k_quant_q(const float* __restrict__ q32, const float* __restrict__ qs,
                          float* __restrict__ out) {
  int i = blockIdx.x * 256 + threadIdx.x;
  if (i >= B_ * NH_ * S_ * D_) return;
  int h = (i / (S_ * D_)) % NH_;
  float v = rintf(q32[i] / qs[h]);
  out[i] = fminf(fmaxf(v, -7.f), 7.f);
}

__global__ void k_quant_k(const float* __restrict__ k32, const float* __restrict__ ks,
                          float* __restrict__ out) {
  int i = blockIdx.x * 256 + threadIdx.x;
  if (i >= B_ * NH_ * S_ * D_) return;
  int d = i % D_;
  int s = (i / D_) % S_;
  int h = (i / (S_ * D_)) % NH_;
  int b = i / (NH_ * S_ * D_);
  int kh = h / NREP_;
  float v = rintf(k32[((size_t)((b * NKV_ + kh) * S_ + s)) * D_ + d] / ks[kh]);
  out[i] = fminf(fmaxf(v, -7.f), 7.f);
}

// ---------------- flash attention (wave32 WMMA + TDM tile staging) ----------------
// grid (S/64, B*NH), 128 threads = 4 waves; each wave owns 16 q rows.
__global__ void __launch_bounds__(128)
k_attn(const _Float16* __restrict__ q16, const _Float16* __restrict__ k16,
       const _Float16* __restrict__ vT16, _Float16* __restrict__ ctx16) {
  __shared__ _Float16 Ks[32 * 128];   // [key][d]
  __shared__ _Float16 Vt[128 * 32];   // [d][key]
  __shared__ _Float16 Pst[4][16 * 32];
  const int tid = threadIdx.x, lane = tid & 31, wv = tid >> 5;
  const int lh = lane >> 4, cg = lane & 15;
  const int q0 = blockIdx.x * 64;
  const int b = blockIdx.y / NH_, h = blockIdx.y % NH_, kh = h / NREP_;

  v8f zero = {};
  v8f acc[8] = {zero, zero, zero, zero, zero, zero, zero, zero};
  float m[8], l[8];
#pragma unroll
  for (int r = 0; r < 8; ++r) { m[r] = -1e30f; l[r] = 0.f; }

  const _Float16* qbase = q16 + ((size_t)((b * NH_ + h) * S_) + q0 + wv * 16) * D_;
  v16h aq[4];
#pragma unroll
  for (int c = 0; c < 4; ++c) aq[c] = frag_a(qbase, D_, c * 32, lane);

  const _Float16* kbase = k16  + ((size_t)(b * NKV_ + kh) * S_) * D_;
  const _Float16* vbase = vT16 + ((size_t)(b * NKV_ + kh) * D_) * S_;
  const int ntiles = 2 * (blockIdx.x + 1);   // causal: keys <= q0+63

  for (int t = 0; t < ntiles; ++t) {
    const int keybase = t * 32;
    __syncthreads();
    if (wv == 0) {
      // K tile: 32 keys x 128 f16 = 32 rows x 64 dwords, tensor row stride 64 dwords
      tdm_load_2d(kbase + (size_t)keybase * D_, (unsigned)(size_t)Ks,
                  64u, (unsigned)S_, 64u, 32u, 64u);
      // V^T tile: 128 d-rows x 32 keys f16 = 128 rows x 16 dwords, stride S/2 dwords
      tdm_load_2d(vbase + keybase, (unsigned)(size_t)Vt,
                  (unsigned)(S_ / 2), (unsigned)D_, 16u, 128u, (unsigned)(S_ / 2));
      __builtin_amdgcn_s_wait_tensorcnt(0);
    }
    __syncthreads();

    v8f s0 = zero, s1 = zero;                        // scores 16 x 32
#pragma unroll
    for (int c = 0; c < 4; ++c) {
      s0 = WMMA_F32F16(aq[c], frag_b(Ks,            128, c * 32, lane), s0);
      s1 = WMMA_F32F16(aq[c], frag_b(Ks + 16 * 128, 128, c * 32, lane), s1);
    }

    float alpha[8], p0[8], p1[8];
#pragma unroll
    for (int r = 0; r < 8; ++r) {
      int qr = q0 + wv * 16 + r + 8 * lh;
      float x0 = s0[r] * SCALING_ + (((keybase + cg)      <= qr) ? 0.f : NEG_INF_);
      float x1 = s1[r] * SCALING_ + (((keybase + 16 + cg) <= qr) ? 0.f : NEG_INF_);
      float tmax = fmaxf(x0, x1);
      for (int mm = 1; mm < 16; mm <<= 1) tmax = fmaxf(tmax, __shfl_xor(tmax, mm, 16));
      float mn = fmaxf(m[r], tmax);
      float a_ = __expf(m[r] - mn);
      float e0 = __expf(x0 - mn), e1 = __expf(x1 - mn);
      float rs = e0 + e1;
      for (int mm = 1; mm < 16; mm <<= 1) rs += __shfl_xor(rs, mm, 16);
      l[r] = l[r] * a_ + rs;
      m[r] = mn; alpha[r] = a_; p0[r] = e0; p1[r] = e1;
    }
#pragma unroll
    for (int j = 0; j < 8; ++j)
#pragma unroll
      for (int r = 0; r < 8; ++r) acc[j][r] *= alpha[r];

    _Float16* Pw = &Pst[wv][0];                      // C-layout -> A-layout via LDS
#pragma unroll
    for (int r = 0; r < 8; ++r) {
      int M = r + 8 * lh;
      Pw[M * 32 + cg]      = (_Float16)p0[r];
      Pw[M * 32 + 16 + cg] = (_Float16)p1[r];
    }
    __syncthreads();
    v16h ap = frag_a(Pw, 32, 0, lane);
#pragma unroll
    for (int j = 0; j < 8; ++j) {
      v16h bv = frag_b(Vt + j * 16 * 32, 32, 0, lane);
      acc[j] = WMMA_F32F16(ap, bv, acc[j]);
    }
  }

#pragma unroll
  for (int j = 0; j < 8; ++j)
#pragma unroll
    for (int r = 0; r < 8; ++r) {
      int row = q0 + wv * 16 + r + 8 * lh;
      ctx16[((size_t)(b * S_ + row) * NH_ + h) * D_ + j * 16 + cg] =
          (_Float16)(acc[j][r] / l[r]);
    }
}

// ---------------- host side ----------------
extern "C" void kernel_launch(void* const* d_in, const int* in_sizes, int n_in,
                              void* d_out, int out_size, void* d_ws, size_t ws_size,
                              hipStream_t stream) {
  (void)in_sizes; (void)n_in; (void)out_size; (void)ws_size;
  const float* hs   = (const float*)d_in[0];
  const float* cosb = (const float*)d_in[1];
  const float* sinb = (const float*)d_in[2];
  // d_in[3] (attention_mask) is exactly causal -> computed analytically in-kernel
  const float* Wq = (const float*)d_in[4];
  const float* Wk = (const float*)d_in[5];
  const float* Wv = (const float*)d_in[6];
  const float* Wo = (const float*)d_in[7];

  char* w = (char*)d_ws;
  auto alloc = [&](size_t bytes) -> void* {
    void* p = (void*)w;
    w += (bytes + 255) & ~(size_t)255;
    return p;
  };
  const size_t NQ  = (size_t)B_ * NH_ * S_ * D_;    // 8388608
  const size_t NKE = (size_t)B_ * NKV_ * S_ * D_;   // 2097152
  const size_t NHS = (size_t)B_ * S_ * HID_;        // 8388608

  _Float16* hs16 = (_Float16*)alloc(NHS * 2);
  _Float16* WqT  = (_Float16*)alloc((size_t)HID_ * NH_ * D_ * 2);
  _Float16* WkT  = (_Float16*)alloc((size_t)HID_ * NKV_ * D_ * 2);
  _Float16* WvT  = (_Float16*)alloc((size_t)HID_ * NKV_ * D_ * 2);
  _Float16* WoT  = (_Float16*)alloc((size_t)NH_ * D_ * HID_ * 2);
  float*    qproj = (float*)alloc(NQ * 4);
  float*    kproj = (float*)alloc(NKE * 4);
  float*    vproj = (float*)alloc(NKE * 4);
  float*    q32   = (float*)alloc(NQ * 4);
  float*    k32   = (float*)alloc(NKE * 4);
  _Float16* q16   = (_Float16*)alloc(NQ * 2);
  _Float16* k16   = (_Float16*)alloc(NKE * 2);
  _Float16* vT16  = (_Float16*)alloc(NKE * 2);
  _Float16* ctx16 = (_Float16*)alloc(NQ * 2);
  unsigned* absq  = (unsigned*)alloc(NH_ * 4);
  unsigned* absk  = (unsigned*)alloc(NKV_ * 4);
  unsigned* histq = (unsigned*)alloc((size_t)NH_ * NB_ * 4);
  unsigned* histk = (unsigned*)alloc((size_t)NKV_ * NB_ * 4);
  float*    qsw   = (float*)alloc(NH_ * 4);
  float*    ksw   = (float*)alloc(NKV_ * 4);

  float* out      = (float*)d_out;
  float* qint_out = out + NHS;
  float* qs_out   = qint_out + NQ;
  float* kint_out = qs_out + NH_;
  float* ks_out   = kint_out + NQ;

  k_cvt16<<<(int)(NHS / 256), 256, 0, stream>>>(hs, hs16, (int)NHS);
  k_cvtT<<<(HID_ * NH_ * D_) / 256, 256, 0, stream>>>(Wq, WqT, HID_, NH_ * D_);
  k_cvtT<<<(HID_ * NKV_ * D_) / 256, 256, 0, stream>>>(Wk, WkT, HID_, NKV_ * D_);
  k_cvtT<<<(HID_ * NKV_ * D_) / 256, 256, 0, stream>>>(Wv, WvT, HID_, NKV_ * D_);
  k_cvtT<<<(NH_ * D_ * HID_) / 256, 256, 0, stream>>>(Wo, WoT, NH_ * D_, HID_);
  k_zero<<<(NH_ * NB_ + 255) / 256, 256, 0, stream>>>(histq, NH_ * NB_);
  k_zero<<<(NKV_ * NB_ + 255) / 256, 256, 0, stream>>>(histk, NKV_ * NB_);
  k_zero<<<1, 32, 0, stream>>>(absq, NH_);
  k_zero<<<1, 32, 0, stream>>>(absk, NKV_);

  k_gemm<<<dim3((NH_ * D_) / 64, (B_ * S_) / 64), 128, 0, stream>>>(hs16, WqT, qproj, B_ * S_, NH_ * D_, HID_);
  k_gemm<<<dim3((NKV_ * D_) / 64, (B_ * S_) / 64), 128, 0, stream>>>(hs16, WkT, kproj, B_ * S_, NKV_ * D_, HID_);
  k_gemm<<<dim3((NKV_ * D_) / 64, (B_ * S_) / 64), 128, 0, stream>>>(hs16, WvT, vproj, B_ * S_, NKV_ * D_, HID_);

  k_rope<<<B_ * NH_ * S_, 128, 0, stream>>>(qproj, cosb, sinb, q32, q16, absq, NH_);
  k_rope<<<B_ * NKV_ * S_, 128, 0, stream>>>(kproj, cosb, sinb, k32, k16, absk, NKV_);
  k_vprep<<<(int)(NKE / 256), 256, 0, stream>>>(vproj, vT16);

  k_hist<<<(int)(NQ / 256), 256, 0, stream>>>(q32, absq, histq, NH_);
  k_hist<<<(int)(NKE / 256), 256, 0, stream>>>(k32, absk, histk, NKV_);
  k_scales<<<1, 32, 0, stream>>>(histq, absq, histk, absk, qsw, ksw, qs_out, ks_out);
  k_quant_q<<<(int)(NQ / 256), 256, 0, stream>>>(q32, qsw, qint_out);
  k_quant_k<<<(int)(NQ / 256), 256, 0, stream>>>(k32, ksw, kint_out);

  k_attn<<<dim3(S_ / 64, B_ * NH_), 128, 0, stream>>>(q16, k16, vT16, ctx16);
  k_gemm<<<dim3(HID_ / 64, (B_ * S_) / 64), 128, 0, stream>>>(ctx16, WoT, out, B_ * S_, HID_, NH_ * D_);
}
